// TGCNConv_84344567759039
// MI455X (gfx1250) — matrix operand order repaired and checked by reading, not compiled, not run on hardware
//
#include <hip/hip_runtime.h>

typedef __attribute__((ext_vector_type(16))) _Float16 v16h;
typedef __attribute__((ext_vector_type(8)))  _Float16 v8h;
typedef __attribute__((ext_vector_type(4)))  _Float16 v4h;
typedef __attribute__((ext_vector_type(8)))  float    v8f;

#define BN_EPS 1e-5f

// ---------------- utility kernels ----------------
__global__ void k_fill(float* __restrict__ p, float v, int n) {
  int i = blockIdx.x * blockDim.x + threadIdx.x;
  if (i < n) p[i] = v;
}

__global__ void k_count_deg(float* __restrict__ deg, const int* __restrict__ row, int E) {
  int e = blockIdx.x * blockDim.x + threadIdx.x;
  if (e < E) atomicAdd(&deg[row[e]], 1.0f);
}

__global__ void k_rsqrt(float* __restrict__ p, int n) {
  int i = blockIdx.x * blockDim.x + threadIdx.x;
  if (i < n) p[i] = rsqrtf(p[i]);
}

// ---- W[K][H] f32  ->  Wt[H][K] f16 (transposed, K-contiguous for B fragments) ----
__global__ void k_cvt_wT(const float* __restrict__ W, _Float16* __restrict__ Wt, int K, int H) {
  int k = blockIdx.x;            // grid = K
  int n = threadIdx.x;           // block = H
  Wt[(unsigned)n * (unsigned)K + (unsigned)k] = (_Float16)W[(unsigned)k * (unsigned)H + (unsigned)n];
}

// ---- act+convert: dst_f16[r][f] = AFFINE ? relu(src*scale+shift) : src ----
template <bool AFFINE>
__global__ void k_cvt_act(const float4* __restrict__ src, v4h* __restrict__ dst,
                          const float4* __restrict__ scale, const float4* __restrict__ shift,
                          int K4) {
  unsigned r = blockIdx.x;                     // grid = N rows
  for (int f = threadIdx.x; f < K4; f += blockDim.x) {
    size_t i = (size_t)r * (unsigned)K4 + (unsigned)f;
    float4 v = src[i];
    if (AFFINE) {
      float4 s = scale[f], sh = shift[f];
      v.x = fmaxf(v.x * s.x + sh.x, 0.0f);
      v.y = fmaxf(v.y * s.y + sh.y, 0.0f);
      v.z = fmaxf(v.z * s.z + sh.z, 0.0f);
      v.w = fmaxf(v.w * s.w + sh.w, 0.0f);
    }
    v4h o;
    o[0] = (_Float16)v.x; o[1] = (_Float16)v.y; o[2] = (_Float16)v.z; o[3] = (_Float16)v.w;
    dst[i] = o;
  }
}

// ---------------- direct-from-L2 WMMA GEMM: C[N,H] = Ah[N,K] @ Wt[H,K]^T ----------------
// No LDS, no barriers: f16 operands are L2-resident (A 25.6MB, Wt 128KB << 192MB L2).
// 8 waves/block, each wave owns a 16x64 output strip (4 accumulators).
// Fragment layouts used by this kernel:
//   A: lane l -> M=l%16, K = hsel*16 + j (contiguous in Ah row)  (hsel=l/16)
//   B: lane l -> N=l%16, K = hsel*16 + j (contiguous in Wt row)
//   C: VGPR i -> M = i + 8*hsel, N = l%16
__global__ __launch_bounds__(256) void gemm_wmma(
    const _Float16* __restrict__ Ah, const _Float16* __restrict__ Wt,
    float* __restrict__ C, int N, int K, int H)
{
  const int lane = threadIdx.x & 31;
  const int wave = threadIdx.x >> 5;
  const int hsel = lane >> 4;
  const int lmod = lane & 15;
  const int rowBase = blockIdx.x * 128 + wave * 16;
  const int col0 = blockIdx.y * 64;

  int arow = rowBase + lmod;
  if (arow >= N) arow = N - 1;                 // clamp loads; stores are guarded
  const _Float16* __restrict__ aRow = Ah + (size_t)((unsigned)arow * (unsigned)K);
  const _Float16* __restrict__ bRow = Wt + (size_t)((unsigned)(col0 + lmod) * (unsigned)K) + hsel * 16;
  const size_t bTile = (size_t)16 * (unsigned)K;

  v8f acc[4] = {};
#pragma unroll 2
  for (int kk = 0; kk < K; kk += 32) {
    v8h a0 = *(const v8h*)(aRow + kk + hsel * 16);
    v8h a1 = *(const v8h*)(aRow + kk + hsel * 16 + 8);
    v16h af;
#pragma unroll
    for (int j = 0; j < 8; ++j) { af[j] = a0[j]; af[j + 8] = a1[j]; }
#pragma unroll
    for (int t = 0; t < 4; ++t) {
      const _Float16* bp = bRow + t * bTile + kk;
      v8h b0 = *(const v8h*)bp;
      v8h b1 = *(const v8h*)(bp + 8);
      v16h bf;
#pragma unroll
      for (int j = 0; j < 8; ++j) { bf[j] = b0[j]; bf[j + 8] = b1[j]; }
      acc[t] = __builtin_amdgcn_wmma_f32_16x16x32_f16(
          false, af, false, bf, (short)0, acc[t], false, false);
    }
  }

#pragma unroll
  for (int t = 0; t < 4; ++t) {
    int gr0 = rowBase + hsel * 8;
#pragma unroll
    for (int i = 0; i < 8; ++i) {
      int gr = gr0 + i;
      if (gr < N)
        C[(size_t)((unsigned)gr * (unsigned)H) + col0 + t * 16 + lmod] = acc[t][i];
    }
  }
}

// ---------------- agg init: self-loop + bias (no atomics) ----------------
__global__ void k_selfloop(const float4* __restrict__ hW, float4* __restrict__ agg,
                           const float4* __restrict__ b4, const float* __restrict__ dis, int H4) {
  unsigned n = blockIdx.x;                     // grid = N
  float d = dis[n];
  float nn = d * d;
  for (int f = threadIdx.x; f < H4; f += blockDim.x) {
    size_t i = (size_t)n * (unsigned)H4 + (unsigned)f;
    float4 v = hW[i];
    float4 bb = b4[f];
    float4 o;
    o.x = bb.x + v.x * nn; o.y = bb.y + v.y * nn;
    o.z = bb.z + v.z * nn; o.w = bb.w + v.w * nn;
    agg[i] = o;
  }
}

// ------- edge scatter: 2 edges/wave grid-stride, prefetch next source row -------
__global__ __launch_bounds__(256) void k_scatter(
    const float* __restrict__ hW, float* __restrict__ agg,
    const int* __restrict__ row, const int* __restrict__ col,
    const float* __restrict__ dis, int E, int H) {
  int gw = (blockIdx.x * blockDim.x + threadIdx.x) >> 5;
  int lane = threadIdx.x & 31;
  int nw = (gridDim.x * blockDim.x) >> 5;
  int H4 = H >> 2;
  for (int e = gw; e < E; e += nw) {
    int en = e + nw;
    if (en < E) {   // warm L2/L0 for next iteration's gather row (global_prefetch_b8)
      const float* pf = hW + (size_t)((unsigned)col[en] * (unsigned)H);
      __builtin_prefetch(pf + lane * 8, 0, 0);
    }
    int r = row[e], c = col[e];
    float nrm = dis[r] * dis[c];
    const float4* src4 = (const float4*)(hW + (size_t)((unsigned)c * (unsigned)H));
    float* dst = agg + (size_t)((unsigned)r * (unsigned)H);
    for (int j = lane; j < H4; j += 32) {
      float4 v = src4[j];
      atomicAdd(&dst[4 * j + 0], v.x * nrm);
      atomicAdd(&dst[4 * j + 1], v.y * nrm);
      atomicAdd(&dst[4 * j + 2], v.z * nrm);
      atomicAdd(&dst[4 * j + 3], v.w * nrm);
    }
  }
}

// ---------------- BatchNorm statistics ----------------
__global__ void k_bn_stats(const float* __restrict__ agg, float* __restrict__ sum,
                           float* __restrict__ sumsq, int N, int H) {
  int f = threadIdx.x;                  // blockDim.x == H
  int nb = gridDim.x;
  int rows_per = (N + nb - 1) / nb;
  int r0 = blockIdx.x * rows_per;
  int r1 = r0 + rows_per; if (r1 > N) r1 = N;
  float s = 0.0f, s2 = 0.0f;
  for (int r = r0; r < r1; ++r) {
    float v = agg[(size_t)((unsigned)r * (unsigned)H) + f];
    s += v; s2 += v * v;
  }
  atomicAdd(&sum[f], s);
  atomicAdd(&sumsq[f], s2);
}

__global__ void k_bn_finalize(const float* __restrict__ sum, const float* __restrict__ sumsq,
                              const float* __restrict__ g, const float* __restrict__ be,
                              float* __restrict__ scale, float* __restrict__ shift, int N) {
  int f = threadIdx.x;
  float inv_n = 1.0f / (float)N;
  float mu  = sum[f] * inv_n;
  float var = sumsq[f] * inv_n - mu * mu;
  if (var < 0.0f) var = 0.0f;
  float sc = g[f] * rsqrtf(var + BN_EPS);
  scale[f] = sc;
  shift[f] = be[f] - mu * sc;
}

// ---------------- bias + relu (final linear epilogue) ----------------
__global__ void k_bias_relu(float* __restrict__ h, const float* __restrict__ b, int H) {
  int f = threadIdx.x;                  // blockDim = H, grid = N
  size_t i = (size_t)blockIdx.x * (unsigned)H + (unsigned)f;
  h[i] = fmaxf(h[i] + b[f], 0.0f);
}

// ---------------- gather + H->1 projection: one wave per output row ----------------
__global__ __launch_bounds__(256) void k_final(
    const float* __restrict__ h, const int* __restrict__ ids,
    const float* __restrict__ Wf, const float* __restrict__ bf,
    float* __restrict__ out, int T, int H) {
  int g = blockIdx.x * blockDim.x + threadIdx.x;
  int t = g >> 5;
  int lane = g & 31;
  if (t >= T) return;
  const float* hp = h + (size_t)((unsigned)ids[t] * (unsigned)H);
  float acc = 0.0f;
  for (int j = lane; j < H; j += 32) acc += hp[j] * Wf[j];
#pragma unroll
  for (int off = 16; off > 0; off >>= 1) acc += __shfl_down(acc, off, 32);
  if (lane == 0) out[t] = acc + bf[0];
}

extern "C" void kernel_launch(void* const* d_in, const int* in_sizes, int n_in,
                              void* d_out, int out_size, void* d_ws, size_t ws_size,
                              hipStream_t stream) {
  const float* x   = (const float*)d_in[0];
  const int*   row = (const int*)d_in[1];           // edge_index[0] = destinations
  const int*   ids = (const int*)d_in[2];
  const float* W1  = (const float*)d_in[3];
  const float* b1  = (const float*)d_in[4];
  const float* W2  = (const float*)d_in[5];
  const float* b2  = (const float*)d_in[6];
  const float* W3  = (const float*)d_in[7];
  const float* b3  = (const float*)d_in[8];
  const float* g1  = (const float*)d_in[9];
  const float* be1 = (const float*)d_in[10];
  const float* g2  = (const float*)d_in[11];
  const float* be2 = (const float*)d_in[12];
  const float* g3  = (const float*)d_in[13];
  const float* be3 = (const float*)d_in[14];
  const float* Wl  = (const float*)d_in[15];
  const float* bl  = (const float*)d_in[16];
  const float* Wf  = (const float*)d_in[17];
  const float* bf  = (const float*)d_in[18];

  const int H = in_sizes[4];            // 256
  const int D = in_sizes[3] / H;        // 128
  const int N = in_sizes[0] / D;        // 50000
  const int E = in_sizes[1] / 2;        // 800000
  const int T = in_sizes[2];            // 10000
  const int* col = row + E;             // edge_index[1] = sources

  // ---- workspace carve (256B aligned) ----
  char* ws = (char*)d_ws;
  size_t off = 0;
  auto take = [&](size_t bytes) -> char* {
    off = (off + 255) & ~(size_t)255;
    char* p = ws + off;
    off += bytes;
    return p;
  };
  float*    dis   = (float*)take((size_t)N * 4);
  float*    sum   = (float*)take((size_t)H * 4);
  float*    sumsq = (float*)take((size_t)H * 4);
  float*    scale = (float*)take((size_t)H * 4);
  float*    shift = (float*)take((size_t)H * 4);
  float*    buf0  = (float*)take((size_t)N * H * 4);      // hW  (51.2MB, L2-resident)
  float*    buf1  = (float*)take((size_t)N * H * 4);      // agg (51.2MB, L2-resident)
  _Float16* Ah    = (_Float16*)take((size_t)N * H * 2);   // f16 activations (25.6MB)
  _Float16* W1t   = (_Float16*)take((size_t)H * D * 2);
  _Float16* W2t   = (_Float16*)take((size_t)H * H * 2);
  _Float16* W3t   = (_Float16*)take((size_t)H * H * 2);
  _Float16* Wlt   = (_Float16*)take((size_t)H * H * 2);
  (void)ws_size; (void)n_in; (void)out_size;

  // ---- symmetric-norm degrees (self loop => deg starts at 1) ----
  k_fill<<<(N + 255) / 256, 256, 0, stream>>>(dis, 1.0f, N);
  k_count_deg<<<(E + 255) / 256, 256, 0, stream>>>(dis, row, E);
  k_rsqrt<<<(N + 255) / 256, 256, 0, stream>>>(dis, N);

  // ---- weights: convert + transpose to f16 [H][K] (tiny, stays in cache) ----
  k_cvt_wT<<<D, H, 0, stream>>>(W1, W1t, D, H);
  k_cvt_wT<<<H, H, 0, stream>>>(W2, W2t, H, H);
  k_cvt_wT<<<H, H, 0, stream>>>(W3, W3t, H, H);
  k_cvt_wT<<<H, H, 0, stream>>>(Wl, Wlt, H, H);

  const _Float16* Wts[3] = {W1t, W2t, W3t};
  const float*    bs[3]  = {b1, b2, b3};
  const float*    gs[3]  = {g1, g2, g3};
  const float*    bes[3] = {be1, be2, be3};

  const int scat_waves  = (E + 1) / 2;                  // 2 edges per wave
  const int scat_blocks = (scat_waves + 7) / 8;

  int Din = D;
  for (int l = 0; l < 3; ++l) {
    // activation convert (fused BN+ReLU of previous layer for l>0)
    if (l == 0)
      k_cvt_act<false><<<N, 64, 0, stream>>>((const float4*)x, (v4h*)Ah,
                                             nullptr, nullptr, Din / 4);
    else
      k_cvt_act<true><<<N, 64, 0, stream>>>((const float4*)buf1, (v4h*)Ah,
                                            (const float4*)scale, (const float4*)shift, Din / 4);
    dim3 grid((N + 127) / 128, H / 64);
    gemm_wmma<<<grid, 256, 0, stream>>>(Ah, Wts[l], buf0, N, Din, H);

    k_selfloop<<<N, 64, 0, stream>>>((const float4*)buf0, (float4*)buf1,
                                     (const float4*)bs[l], dis, H / 4);
    k_scatter<<<scat_blocks, 256, 0, stream>>>(buf0, buf1, row, col, dis, E, H);

    k_fill<<<(H + 255) / 256, 256, 0, stream>>>(sum, 0.0f, H);
    k_fill<<<(H + 255) / 256, 256, 0, stream>>>(sumsq, 0.0f, H);
    k_bn_stats<<<128, H, 0, stream>>>(buf1, sum, sumsq, N, H);
    k_bn_finalize<<<1, H, 0, stream>>>(sum, sumsq, gs[l], bes[l], scale, shift, N);
    Din = H;
  }

  // ---- h = relu(relu(bn3(agg3)) @ Wl + bl) ----
  k_cvt_act<true><<<N, 64, 0, stream>>>((const float4*)buf1, (v4h*)Ah,
                                        (const float4*)scale, (const float4*)shift, H / 4);
  {
    dim3 grid((N + 127) / 128, H / 64);
    gemm_wmma<<<grid, 256, 0, stream>>>(Ah, Wlt, buf0, N, H, H);
  }
  k_bias_relu<<<N, H, 0, stream>>>(buf0, bl, H);

  // ---- gather train ids + final H->1 projection ----
  k_final<<<(T * 32 + 255) / 256, 256, 0, stream>>>(buf0, ids, Wf, bf, (float*)d_out, T, H);
}
